// GraphRegularized_38439957299681
// MI455X (gfx1250) — compile-verified, matrix operand order checked by reading, not run on hardware
//
#include <hip/hip_runtime.h>
#include <hip/hip_bf16.h>

// ---------------------------------------------------------------------------
// Graph-regularized loss on MI455X (gfx1250):
//   loss = sum_{i!=j} [||x_i-x_j|| <= 46] * exp(-||x_i-x_j||/10) * ||y_i-y_j||
// Dual fused Gram matrices via bf16 hi/lo split WMMA (3 products per K-chunk
// ~ f32 accuracy). LDS staging uses gfx1250 async loads
// (global_load_async_to_lds_b128, ASYNCcnt) with double buffering so the
// fill of chunk c+1 overlaps the WMMA burst of chunk c. Distance matrices
// never touch memory; single-scalar atomic reduction.
// ---------------------------------------------------------------------------

typedef __attribute__((ext_vector_type(16))) __bf16 v16bf;
typedef __attribute__((ext_vector_type(8)))  __bf16 v8bf;
typedef __attribute__((ext_vector_type(8)))  float  v8f;

union ABOp { v16bf v; v8bf h[2]; };

#define B_N     4096
#define K_X     1024
#define K_Y     512
#define EPS_THR 46.0f
#define INV_SIG 0.1f

// ---------------------------------------------------------------------------
// Prep: split f32 rows into bf16 hi/lo pair and compute row squared norms.
// ---------------------------------------------------------------------------
__global__ __launch_bounds__(256) void prep_rows_kernel(
    const float* __restrict__ src, __bf16* __restrict__ hi,
    __bf16* __restrict__ lo, float* __restrict__ sq, int ncols)
{
    const int row = blockIdx.x;
    const size_t base = (size_t)row * (size_t)ncols;
    float s = 0.0f;
    for (int c = threadIdx.x; c < ncols; c += 256) {
        const float v = src[base + c];
        const __bf16 h = (__bf16)v;                 // RNE
        const __bf16 l = (__bf16)(v - (float)h);    // residual
        hi[base + c] = h;
        lo[base + c] = l;
        s += v * v;
    }
    __shared__ float red[256];
    red[threadIdx.x] = s;
    __syncthreads();
#pragma unroll
    for (int st = 128; st > 0; st >>= 1) {
        if (threadIdx.x < st) red[threadIdx.x] += red[threadIdx.x + st];
        __syncthreads();
    }
    if (threadIdx.x == 0) sq[row] = red[0];
}

__global__ void zero_kernel(float* __restrict__ p, int n)
{
    const int i = blockIdx.x * blockDim.x + threadIdx.x;
    if (i < n) p[i] = 0.0f;
}

// ---------------------------------------------------------------------------
// Main fused kernel: each block owns a 128x128 region of the BxB matrix.
// 8 waves; wave w owns i-tile w (16 rows), sweeps 8 j-tiles.
// LDS double buffer: 2 x {Ahi,Alo,Bhi,Blo} x 128 rows x 32 K = 64 KB.
// ---------------------------------------------------------------------------
__global__ __launch_bounds__(256) void gram_loss_kernel(
    const __bf16* __restrict__ xhi, const __bf16* __restrict__ xlo,
    const __bf16* __restrict__ yhi, const __bf16* __restrict__ ylo,
    const float* __restrict__ sqx, const float* __restrict__ sqy,
    float* __restrict__ out)
{
    __shared__ __bf16 sAhi[2][128 * 32];
    __shared__ __bf16 sAlo[2][128 * 32];
    __shared__ __bf16 sBhi[2][128 * 32];
    __shared__ __bf16 sBlo[2][128 * 32];

    const int t      = threadIdx.x;
    const int wave   = t >> 5;
    const int lane   = t & 31;
    const int laneHi = lane >> 4;   // 0: lanes 0-15, 1: lanes 16-31
    const int laneLo = lane & 15;
    const int iBase  = blockIdx.y * 128;
    const int jBase  = blockIdx.x * 128;

    const int  fillRow = t & 127;
    const bool fillB   = (t >= 128);

    v8f accx[8], accy[8];
    const v8f vzero = {};
#pragma unroll
    for (int jt = 0; jt < 8; ++jt) { accx[jt] = vzero; accy[jt] = vzero; }

    // Issue async memory->LDS fill of one 32-wide K chunk into buffer `buf`.
    // 256 threads x (4 hi + 4 lo) x 16B = 32 KB, no data VGPRs, ASYNCcnt.
    auto issueChunk = [&](const __bf16* __restrict__ ghi,
                          const __bf16* __restrict__ glo,
                          int K, int kc, int buf) {
        const size_t grow = (size_t)((fillB ? jBase : iBase) + fillRow);
        const __bf16* srcH = ghi + grow * (size_t)K + kc;
        const __bf16* srcL = glo + grow * (size_t)K + kc;
        __bf16* dstH = (fillB ? sBhi[buf] : sAhi[buf]) + fillRow * 32;
        __bf16* dstL = (fillB ? sBlo[buf] : sAlo[buf]) + fillRow * 32;
#pragma unroll
        for (int q = 0; q < 4; ++q) {
            // LDS byte offset = low 32 bits of the generic shared address.
            unsigned lH = (unsigned)(size_t)(dstH + q * 8);
            unsigned lL = (unsigned)(size_t)(dstL + q * 8);
            unsigned long long gH = (unsigned long long)(size_t)(srcH + q * 8);
            unsigned long long gL = (unsigned long long)(size_t)(srcL + q * 8);
            asm volatile("global_load_async_to_lds_b128 %0, %1, off"
                         :: "v"(lH), "v"(gH) : "memory");
            asm volatile("global_load_async_to_lds_b128 %0, %1, off"
                         :: "v"(lL), "v"(gL) : "memory");
        }
    };

    // WMMA burst over one staged chunk.
    auto computeChunk = [&](int buf, v8f (&acc)[8]) {
        // ISA 16-bit A layout: lane L: M = L%16; halves 0-7 -> K = h+8*(L/16),
        // halves 8-15 -> K = (h-8)+16+8*(L/16) => two contiguous 16B runs.
        ABOp ahi, alo;
        const __bf16* pa = sAhi[buf] + (wave * 16 + laneLo) * 32 + laneHi * 8;
        const __bf16* qa = sAlo[buf] + (wave * 16 + laneLo) * 32 + laneHi * 8;
        ahi.h[0] = *(const v8bf*)(pa);
        ahi.h[1] = *(const v8bf*)(pa + 16);
        alo.h[0] = *(const v8bf*)(qa);
        alo.h[1] = *(const v8bf*)(qa + 16);
#pragma unroll
        for (int jt = 0; jt < 8; ++jt) {
            // B layout mirrors A (lane <-> N) for the Gram product.
            ABOp bhi, blo;
            const __bf16* pb = sBhi[buf] + (jt * 16 + laneLo) * 32 + laneHi * 8;
            const __bf16* qb = sBlo[buf] + (jt * 16 + laneLo) * 32 + laneHi * 8;
            bhi.h[0] = *(const v8bf*)(pb);
            bhi.h[1] = *(const v8bf*)(pb + 16);
            blo.h[0] = *(const v8bf*)(qb);
            blo.h[1] = *(const v8bf*)(qb + 16);

            v8f c = acc[jt];
            // dot ~= hi*hi + hi*lo + lo*hi  (bf16x3 split)
            c = __builtin_amdgcn_wmma_f32_16x16x32_bf16(
                    false, ahi.v, false, bhi.v, (short)0, c, false, false);
            c = __builtin_amdgcn_wmma_f32_16x16x32_bf16(
                    false, ahi.v, false, blo.v, (short)0, c, false, false);
            c = __builtin_amdgcn_wmma_f32_16x16x32_bf16(
                    false, alo.v, false, bhi.v, (short)0, c, false, false);
            acc[jt] = c;
        }
    };

    // Double-buffered pipeline: async fill of chunk c+1 overlaps WMMA on c.
    auto phase = [&](const __bf16* __restrict__ ghi,
                     const __bf16* __restrict__ glo,
                     int K, v8f (&acc)[8]) {
        const int nchunks = K / 32;
        issueChunk(ghi, glo, K, 0, 0);
        for (int c = 0; c < nchunks; ++c) {
            const int cur = c & 1;
            // my in-flight async writes done -> barrier: data globally visible
            // and the buffer we are about to overwrite has no readers left.
            asm volatile("s_wait_asynccnt 0" ::: "memory");
            __syncthreads();
            if (c + 1 < nchunks)
                issueChunk(ghi, glo, K, (c + 1) * 32, cur ^ 1);
            computeChunk(cur, acc);
        }
    };

    phase(xhi, xlo, K_X, accx);
    phase(yhi, ylo, K_Y, accy);

    // Fused epilogue. C/D layout: element (vgpr r, lane L) is
    // M = r + 8*(L/16), N = L%16.
    float partial = 0.0f;
    const int iRow0 = iBase + wave * 16 + 8 * laneHi;
    float sxi[8], syi[8];
#pragma unroll
    for (int r = 0; r < 8; ++r) {
        sxi[r] = sqx[iRow0 + r];
        syi[r] = sqy[iRow0 + r];
    }
#pragma unroll
    for (int jt = 0; jt < 8; ++jt) {
        const int j   = jBase + jt * 16 + laneLo;
        const float sxj = sqx[j];
        const float syj = sqy[j];
#pragma unroll
        for (int r = 0; r < 8; ++r) {
            const int i = iRow0 + r;
            float d2x = sxi[r] + sxj - 2.0f * accx[jt][r];
            d2x = fmaxf(d2x, 0.0f);
            const float dxv = sqrtf(d2x > 1e-12f ? d2x : 1e-12f);
            float d2y = syi[r] + syj - 2.0f * accy[jt][r];
            d2y = fmaxf(d2y, 0.0f);
            const float dyv = sqrtf(d2y > 1e-12f ? d2y : 1e-12f);
            if ((i != j) && (dxv <= EPS_THR))
                partial += __expf(-dxv * INV_SIG) * dyv;
        }
    }

    // wave32 reduce, one atomic per wave.
#pragma unroll
    for (int s = 16; s > 0; s >>= 1)
        partial += __shfl_down(partial, s, 32);
    if (lane == 0) atomicAdd(out, partial);
}

// ---------------------------------------------------------------------------
// Launch: d_in[0] = x [4096,1024] f32, d_in[1] = y [4096,512] f32.
// Workspace layout (24 MiB + 32 KiB):
//   [0,8M):   xhi   [8M,16M): xlo   [16M,20M): yhi   [20M,24M): ylo
//   [24M):    sqx (16KB)     then sqy (16KB)
// ---------------------------------------------------------------------------
extern "C" void kernel_launch(void* const* d_in, const int* in_sizes, int n_in,
                              void* d_out, int out_size, void* d_ws, size_t ws_size,
                              hipStream_t stream)
{
    const float* x = (const float*)d_in[0];
    const float* y = (const float*)d_in[1];
    float* out = (float*)d_out;
    char* ws = (char*)d_ws;

    const size_t MB = 1024u * 1024u;
    __bf16* xhi = (__bf16*)(ws);
    __bf16* xlo = (__bf16*)(ws + 8 * MB);
    __bf16* yhi = (__bf16*)(ws + 16 * MB);
    __bf16* ylo = (__bf16*)(ws + 20 * MB);
    float*  sqx = (float*)(ws + 24 * MB);
    float*  sqy = (float*)(ws + 24 * MB + 16384u);

    zero_kernel<<<1, 64, 0, stream>>>(out, out_size);
    prep_rows_kernel<<<B_N, 256, 0, stream>>>(x, xhi, xlo, sqx, K_X);
    prep_rows_kernel<<<B_N, 256, 0, stream>>>(y, yhi, ylo, sqy, K_Y);

    dim3 grid(B_N / 128, B_N / 128, 1);
    gram_loss_kernel<<<grid, 256, 0, stream>>>(xhi, xlo, yhi, ylo, sqx, sqy, out);
}
